// NoiseFilter_53730040873034
// MI455X (gfx1250) — compile-verified, matrix operand order checked by reading, not run on hardware
//
#include <hip/hip_runtime.h>
#include <hip/hip_bf16.h>

// ---------------------------------------------------------------------------
// CDNA5 (gfx1250). wave32; v_wmma_f32_16x16x32_f16 GEMM core.
//   - adj_hat2 never materialized:  D A2 D v = d .* (A2 (d .* v))
//   - adj2 stored f16 (halves HBM traffic of the dominant GC passes)
//   - f16-A tiles staged with GLOBAL_LOAD_ASYNC_TO_LDS_B128 when available
//   - split-K (gridDim.z>1) via global_atomic_add_f32 for occupancy on the
//     N<=32, K=4096 bandwidth-bound passes; epilogue applied in-place after
// ---------------------------------------------------------------------------

typedef __attribute__((ext_vector_type(16))) _Float16 v16h;
typedef __attribute__((ext_vector_type(8)))  _Float16 v8h;
typedef __attribute__((ext_vector_type(4)))  _Float16 v4h;
typedef __attribute__((ext_vector_type(8)))  float    v8f;
typedef __attribute__((ext_vector_type(4)))  float    v4f;
typedef int v4i_gnu __attribute__((vector_size(16)));   // matches builtin's V4i

#define BM 128
#define BN 32
#define BK 32
#define LDS_STRIDE 40   // halfs; 80B row stride -> conflict-free b128 LDS ops

#if defined(__AMDGCN__) && __has_builtin(__builtin_amdgcn_global_load_async_to_lds_b128) && __has_builtin(__builtin_amdgcn_s_wait_asynccnt)
#define HAVE_ASYNC_LDS 1
#else
#define HAVE_ASYNC_LDS 0
#endif

#if HAVE_ASYNC_LDS
__device__ __forceinline__ void async_copy16(const void* g, void* l)
{
    __builtin_amdgcn_global_load_async_to_lds_b128(
        (__attribute__((address_space(1))) v4i_gnu*)g,
        (__attribute__((address_space(3))) v4i_gnu*)l, 0, 0);
}
#endif

template <typename AT>   // AT = float (convert at staging) or _Float16 (async copy)
__global__ __launch_bounds__(256)
void gemm_wmma(const AT* __restrict__ A, int lda,
               const float* __restrict__ B, int ldb,
               float* __restrict__ C,
               int M, int N, int K,
               const float* __restrict__ bias,
               const float* __restrict__ rowscale,
               int flags)   // bit0 = ReLU, bit1 = B is [N][K] (transposed access)
{
    __shared__ _Float16 sA[BM * LDS_STRIDE];
    __shared__ _Float16 sB[BN * LDS_STRIDE];

    const int tid  = threadIdx.x;
    const int wave = tid >> 5;
    const int lane = tid & 31;
    const int half = lane >> 4;
    const int l16  = lane & 15;

    const int mBase = blockIdx.y * BM;
    const int nBase = blockIdx.x * BN;
    const bool fullN = (nBase + BN) <= N;

    const int kLen = K / gridDim.z;      // split-K range
    const int k0   = blockIdx.z * kLen;
    const int nk   = kLen / BK;

    v8f c0 = {};
    v8f c1 = {};

    for (int kt = 0; kt < nk; ++kt) {
        const int kOff = k0 + kt * BK;

        // ---- stage A tile (BM x BK) into LDS ----
        if constexpr (sizeof(AT) == 2) {
            // f16 source: byte-exact 16B chunks; 512 chunks, 2 per thread
            #pragma unroll
            for (int qq = 0; qq < 2; ++qq) {
                const int q = tid + qq * 256;
                const int r = q >> 2, cchunk = q & 3;
                const AT* gp = A + (size_t)(mBase + r) * lda + kOff + cchunk * 8;
                _Float16* lp = &sA[r * LDS_STRIDE + cchunk * 8];
#if HAVE_ASYNC_LDS
                async_copy16(gp, lp);
#else
                *(v8h*)lp = *(const v8h*)gp;
#endif
            }
        } else {
            // f32 source: load + v_cvt_pk_f16_f32 + b128 LDS store
            const int r  = tid >> 1;
            const int cc = (tid & 1) * 16;
            const AT* ap = A + (size_t)(mBase + r) * lda + kOff + cc;
            _Float16* dst = &sA[r * LDS_STRIDE + cc];
            #pragma unroll
            for (int i = 0; i < 16; ++i) dst[i] = (_Float16)ap[i];
        }

        // ---- stage B tile transposed into sB[n][k]; 4 elems / thread ----
        {
            const int t4 = tid << 2;
            if (flags & 2) {
                // B stored [N][K]: contiguous along k
                const int n = t4 >> 5, k = t4 & 31;
                const int gn = nBase + n;
                v4f v = {};
                if (gn < N) v = *(const v4f*)(B + (size_t)gn * ldb + kOff + k);
                v4h p;
                #pragma unroll
                for (int i = 0; i < 4; ++i) p[i] = (_Float16)v[i];
                *(v4h*)&sB[n * LDS_STRIDE + k] = p;
            } else {
                // B stored [K][N]: contiguous along n
                const int k = t4 >> 5, n = t4 & 31;
                const float* bp = B + (size_t)(kOff + k) * ldb + nBase + n;
                if (fullN) {
                    v4f v = *(const v4f*)bp;
                    #pragma unroll
                    for (int i = 0; i < 4; ++i)
                        sB[(n + i) * LDS_STRIDE + k] = (_Float16)v[i];
                } else {
                    #pragma unroll
                    for (int i = 0; i < 4; ++i) {
                        const float v = ((nBase + n + i) < N) ? bp[i] : 0.f;
                        sB[(n + i) * LDS_STRIDE + k] = (_Float16)v;
                    }
                }
            }
        }

#if HAVE_ASYNC_LDS
        if constexpr (sizeof(AT) == 2) __builtin_amdgcn_s_wait_asynccnt(0);
#endif
        __syncthreads();

        if (kt + 1 < nk) {
            const AT* pf = A + (size_t)(mBase + (tid >> 1)) * lda + kOff + BK;
            __builtin_prefetch(pf, 0, 1);
        }

        // ---- fragments per ISA 7.12.2: element e -> k = (e/8)*16+8*half+(e%8)
        const int arow = wave * 16 + l16;
        v8h alo = *(const v8h*)&sA[arow * LDS_STRIDE + half * 8];
        v8h ahi = *(const v8h*)&sA[arow * LDS_STRIDE + 16 + half * 8];
        v16h a;
        #pragma unroll
        for (int e = 0; e < 8; ++e) { a[e] = alo[e]; a[8 + e] = ahi[e]; }

        v8h b0lo = *(const v8h*)&sB[(l16)      * LDS_STRIDE + half * 8];
        v8h b0hi = *(const v8h*)&sB[(l16)      * LDS_STRIDE + 16 + half * 8];
        v8h b1lo = *(const v8h*)&sB[(16 + l16) * LDS_STRIDE + half * 8];
        v8h b1hi = *(const v8h*)&sB[(16 + l16) * LDS_STRIDE + 16 + half * 8];
        v16h b0, b1;
        #pragma unroll
        for (int e = 0; e < 8; ++e) {
            b0[e] = b0lo[e]; b0[8 + e] = b0hi[e];
            b1[e] = b1lo[e]; b1[8 + e] = b1hi[e];
        }

        c0 = __builtin_amdgcn_wmma_f32_16x16x32_f16(false, a, false, b0,
                                                    (short)0, c0, false, false);
        c1 = __builtin_amdgcn_wmma_f32_16x16x32_f16(false, a, false, b1,
                                                    (short)0, c1, false, false);
        __syncthreads();
    }

    // ---- writeback: C/D layout -> m = r + 8*half, n = l16 per 16x16 tile
    const int mRow = mBase + wave * 16 + 8 * half;
    if (gridDim.z == 1) {
        #pragma unroll
        for (int r = 0; r < 8; ++r) {
            const int m = mRow + r;
            const float rs = rowscale ? rowscale[m] : 1.f;
            #pragma unroll
            for (int j = 0; j < 2; ++j) {
                const int n = nBase + j * 16 + l16;
                if (n < N) {
                    float v = (j == 0) ? c0[r] : c1[r];
                    v *= rs;
                    if (bias) v += bias[n];
                    if (flags & 1) v = fmaxf(v, 0.f);
                    C[(size_t)m * N + n] = v;
                }
            }
        }
    } else {
        #pragma unroll
        for (int r = 0; r < 8; ++r) {
            const int m = mRow + r;
            #pragma unroll
            for (int j = 0; j < 2; ++j) {
                const int n = nBase + j * 16 + l16;
                if (n < N) atomicAdd(&C[(size_t)m * N + n], (j == 0) ? c0[r] : c1[r]);
            }
        }
    }
}

// in-place epilogue for split-K outputs
__global__ void epilogue_kernel(float* __restrict__ C, int M, int N,
                                const float* __restrict__ bias,
                                const float* __restrict__ rowscale, int relu)
{
    const int i = blockIdx.x * blockDim.x + threadIdx.x;
    if (i < M * N) {
        const int m = i / N, n = i % N;
        float v = C[i];
        if (rowscale) v *= rowscale[m];
        if (bias) v += bias[n];
        if (relu) v = fmaxf(v, 0.f);
        C[i] = v;
    }
}

__global__ void zero_kernel(float* __restrict__ p, int n)
{
    const int i = blockIdx.x * blockDim.x + threadIdx.x;
    if (i < n) p[i] = 0.f;
}

// one block per row: a2h = f16(exp(s - rowmax)), dvec = rsqrt(rowsum)
__global__ __launch_bounds__(256)
void row_expnorm(const float* __restrict__ s, _Float16* __restrict__ a2h,
                 float* __restrict__ dvec)
{
    __shared__ float red[256];
    const int tid = threadIdx.x;
    const float* p = s + (size_t)blockIdx.x * 4096;
    _Float16* q = a2h + (size_t)blockIdx.x * 4096;

    float vals[16];
    float mx = -3.4e38f;
    #pragma unroll
    for (int i = 0; i < 16; ++i) { vals[i] = p[tid + i * 256]; mx = fmaxf(mx, vals[i]); }
    red[tid] = mx; __syncthreads();
    for (int st = 128; st > 0; st >>= 1) {
        if (tid < st) red[tid] = fmaxf(red[tid], red[tid + st]);
        __syncthreads();
    }
    mx = red[0]; __syncthreads();

    float sum = 0.f;
    #pragma unroll
    for (int i = 0; i < 16; ++i) {
        const float e = __expf(vals[i] - mx);
        q[tid + i * 256] = (_Float16)e;
        sum += e;
    }
    red[tid] = sum; __syncthreads();
    for (int st = 128; st > 0; st >>= 1) {
        if (tid < st) red[tid] += red[tid + st];
        __syncthreads();
    }
    if (tid == 0) dvec[blockIdx.x] = rsqrtf(red[0]);
}

__global__ void rowscale_kernel(float* __restrict__ out, const float* __restrict__ in,
                                const float* __restrict__ d, int N, int total)
{
    const int i = blockIdx.x * blockDim.x + threadIdx.x;
    if (i < total) out[i] = in[i] * d[i / N];
}

__global__ void combine_kernel(float* __restrict__ out, const float* __restrict__ a,
                               const float* __restrict__ b, int total)
{
    const int i = blockIdx.x * blockDim.x + threadIdx.x;
    if (i < total) out[i] = 0.5f * (a[i] + b[i]);
}

// ---------------------------------------------------------------------------

template <typename AT>
static inline void gemm(hipStream_t st, const AT* A, int lda,
                        const float* B, int ldb, float* C,
                        int M, int N, int K,
                        const float* bias, const float* rs, int flags, int splits)
{
    dim3 grid((N + BN - 1) / BN, M / BM, splits);
    gemm_wmma<AT><<<grid, dim3(256), 0, st>>>(A, lda, B, ldb, C, M, N, K, bias, rs, flags);
}

extern "C" void kernel_launch(void* const* d_in, const int* in_sizes, int n_in,
                              void* d_out, int out_size, void* d_ws, size_t ws_size,
                              hipStream_t stream)
{
    const float* x   = (const float*)d_in[0];   // [4096,1024]
    const float* adj = (const float*)d_in[1];   // [4096,4096]
    const float* W1  = (const float*)d_in[2];   const float* b1  = (const float*)d_in[3];
    const float* W2  = (const float*)d_in[4];   const float* b2  = (const float*)d_in[5];
    const float* Wg1 = (const float*)d_in[6];   const float* bg1 = (const float*)d_in[7];
    const float* Wg2 = (const float*)d_in[8];   const float* bg2 = (const float*)d_in[9];
    const float* Wg3 = (const float*)d_in[10];  const float* bg3 = (const float*)d_in[11];
    const float* Wg4 = (const float*)d_in[12];  const float* bg4 = (const float*)d_in[13];
    float* out = (float*)d_out;                 // [4096,2]

    const int N = 4096;
    float* ws = (float*)d_ws;
    float* h1  = ws;                    // 4096*512
    float* h   = h1  + (size_t)N * 512; // 4096*128
    float* s   = h   + (size_t)N * 128; // 4096*4096 f32 similarities
    float* t1  = s   + (size_t)N * N;   // 4096*32
    float* t3  = t1  + (size_t)N * 32;
    float* t3s = t3  + (size_t)N * 32;
    float* x1a = t3s + (size_t)N * 32;
    float* y2a = x1a + (size_t)N * 32;
    float* u1  = y2a + (size_t)N * 32;  // 4096*2
    float* u2  = u1  + (size_t)N * 2;
    float* u2s = u2  + (size_t)N * 2;
    float* x1v = u2s + (size_t)N * 2;
    float* y2v = x1v + (size_t)N * 2;
    float* dv  = y2v + (size_t)N * 2;   // 4096
    _Float16* a2h = (_Float16*)(dv + N); // 4096*4096 f16 adj2

    const int EB = 256;
    #define CEIL(a) (((a) + EB - 1) / EB)

    // feature MLP
    gemm(stream, x,  1024, W1, 512, h1, N, 512, 1024, b1, nullptr, 1, 1); // relu
    gemm(stream, h1, 512,  W2, 128, h,  N, 128, 512,  b2, nullptr, 1, 1); // relu

    // tiny projections of h
    gemm(stream, h, 128, Wg1, 32, t1, N, 32, 128, nullptr, nullptr, 0, 1);
    gemm(stream, h, 128, Wg3, 32, t3, N, 32, 128, nullptr, nullptr, 0, 1);

    // similarity matrix s = h @ h^T  (B transposed access)
    gemm(stream, h, 128, h, 128, s, N, N, 128, nullptr, nullptr, 2, 1);

    // adj2(f16) = exp(s - rowmax); dv = rowsum^-0.5
    row_expnorm<<<N, 256, 0, stream>>>(s, a2h, dv);
    rowscale_kernel<<<CEIL(N * 32), EB, 0, stream>>>(t3s, t3, dv, 32, N * 32);

    // branch 1:  x1a = relu(adj @ t1 + bg1)   (split-K, f32 A)
    zero_kernel<<<CEIL(N * 32), EB, 0, stream>>>(x1a, N * 32);
    gemm(stream, adj, N, t1, 32, x1a, N, 32, N, nullptr, nullptr, 0, 16);
    epilogue_kernel<<<CEIL(N * 32), EB, 0, stream>>>(x1a, N, 32, bg1, nullptr, 1);

    // branch 2:  y2a = relu(d .* (adj2 @ (d .* t3)) + bg3)   (split-K, f16 A)
    zero_kernel<<<CEIL(N * 32), EB, 0, stream>>>(y2a, N * 32);
    gemm(stream, (const _Float16*)a2h, N, t3s, 32, y2a, N, 32, N, nullptr, nullptr, 0, 16);
    epilogue_kernel<<<CEIL(N * 32), EB, 0, stream>>>(y2a, N, 32, bg3, dv, 1);

    gemm(stream, x1a, 32, Wg2, 2, u1, N, 2, 32, nullptr, nullptr, 0, 1);
    gemm(stream, y2a, 32, Wg4, 2, u2, N, 2, 32, nullptr, nullptr, 0, 1);
    rowscale_kernel<<<CEIL(N * 2), EB, 0, stream>>>(u2s, u2, dv, 2, N * 2);

    zero_kernel<<<CEIL(N * 2), EB, 0, stream>>>(x1v, N * 2);
    gemm(stream, adj, N, u1, 2, x1v, N, 2, N, nullptr, nullptr, 0, 16);
    epilogue_kernel<<<CEIL(N * 2), EB, 0, stream>>>(x1v, N, 2, bg2, nullptr, 0);

    zero_kernel<<<CEIL(N * 2), EB, 0, stream>>>(y2v, N * 2);
    gemm(stream, (const _Float16*)a2h, N, u2s, 2, y2v, N, 2, N, nullptr, nullptr, 0, 16);
    epilogue_kernel<<<CEIL(N * 2), EB, 0, stream>>>(y2v, N, 2, bg4, dv, 0);

    combine_kernel<<<CEIL(N * 2), EB, 0, stream>>>(out, x1v, y2v, N * 2);
    #undef CEIL
}
